// FFM_19370302505213
// MI455X (gfx1250) — matrix operand order repaired and checked by reading, not compiled, not run on hardware
//
#include <hip/hip_runtime.h>
#include <hip/hip_bf16.h>
#include <math.h>

// ---------------------------------------------------------------------------
// Types for CDNA5 WMMA (wave32, 16x16x32 bf16 -> f32) and TDM descriptors
// ---------------------------------------------------------------------------
typedef __attribute__((ext_vector_type(16))) __bf16        bf16x16;
typedef __attribute__((ext_vector_type(8)))  float         floatx8;
typedef __attribute__((ext_vector_type(4)))  unsigned int  uint4v;
typedef __attribute__((ext_vector_type(4)))  unsigned int  v4u;
typedef __attribute__((ext_vector_type(8)))  int           v8i;
typedef __attribute__((ext_vector_type(4)))  int           v4i;

#define HW   16384   // 128*128
#define IMW  128
#define NB   4       // batch

// f32 -> bf16 bits, round-to-nearest-even
static __device__ __forceinline__ unsigned short f2bf_bits(float f) {
    union { float f; unsigned u; } in; in.f = f;
    unsigned r = in.u + 0x7FFFu + ((in.u >> 16) & 1u);
    return (unsigned short)(r >> 16);
}
static __device__ __forceinline__ __bf16 f2bf(float f) {
    union { unsigned short s; __bf16 b; } out;
    out.s = f2bf_bits(f);
    return out.b;
}

// exact GELU (reference uses approximate=False)
static __device__ __forceinline__ float gelu_exact(float x) {
    return 0.5f * x * (1.0f + erff(x * 0.7071067811865475f));
}

// A-fragment K index for 16-bit 16x32 A layout (lane half selects K ranges)
static __device__ __forceinline__ int a_k(int i, int half) {
    return half * 8 + i + ((i >= 8) ? 8 : 0);  // {h*8..h*8+7} U {16+h*8..16+h*8+7}
}

// ---------------------------------------------------------------------------
// Pointwise conv as WMMA GEMM:  Y[n][co][p] = bias[co] + sum_ci W[co][ci]*X[n][ci][p]
// Block: 256 thr = 8 waves. Tile: 128(M=co) x 64(N=pix) x 32(K=ci).
// Activation tiles are DMA'd by the Tensor Data Mover into a double-buffered
// LDS tile (tensor_load_to_lds + s_wait_tensorcnt pipelining); weights are
// staged fragment-major in bf16 so the A fragment is 2x ds_load_b128.
// Optional fused epilogue: BN (eval) + exact GELU.
// ---------------------------------------------------------------------------
#define SW_LD 40   // bf16 row stride: 80B rows -> 16B aligned, conflict-free b128

template <bool FUSE>
__global__ __launch_bounds__(256)
void pw_gemm(const float* __restrict__ X, long xBS,
             const float* __restrict__ W, const float* __restrict__ bias, int Cin,
             const float* __restrict__ bn_g, const float* __restrict__ bn_b,
             const float* __restrict__ bn_m, const float* __restrict__ bn_v,
             float* __restrict__ Y, long yBS, int yCoff)
{
    __shared__ __attribute__((aligned(16))) __bf16 sW[128 * SW_LD]; // frag-major
    __shared__ __attribute__((aligned(16))) float  sXf[2][32 * 64]; // TDM dest

    const int n   = blockIdx.z;
    const int co0 = blockIdx.y * 128;
    const int p0  = blockIdx.x * 64;
    const int t    = threadIdx.x;
    const int lane = t & 31, wave = t >> 5;
    const int half = lane >> 4, l15 = lane & 15;

    floatx8 acc[4];
    #pragma unroll
    for (int nn = 0; nn < 4; ++nn)
        #pragma unroll
        for (int i = 0; i < 8; ++i) acc[nn][i] = 0.0f;

    const float* Xtile0 = X + (long)n * xBS + p0;   // + k0*HW per step
    // low 32 bits of a generic LDS pointer == logical LDS byte offset
    const unsigned ldsX0 = (unsigned)(size_t)(&sXf[0][0]);
    const unsigned ldsX1 = (unsigned)(size_t)(&sXf[1][0]);

    // TDM: 2D tile, dim0 = 64 pixels (contiguous), dim1 = 32 channels,
    // row stride = HW elements, data_size = 4B. Descriptor per CDNA5 ISA.
    auto tdm_issue = [&](int k0, unsigned ldsOff) {
        const unsigned long long ga =
            (unsigned long long)(size_t)(Xtile0 + (long)k0 * HW);
        v4u g0;
        g0[0] = 1u;                                   // count=1, user mode
        g0[1] = ldsOff;                               // lds_addr
        g0[2] = (unsigned)ga;                         // global_addr[31:0]
        g0[3] = (unsigned)((ga >> 32) & 0x01FFFFFFu)  // global_addr[56:32]
              | 0x80000000u;                          // type=2 (bits 127:126)
        v8i g1;
        g1[0] = 0x20000;                    // data_size=2 (4B), no multicast
        g1[1] = (int)(16384u << 16);        // tensor_dim0 = HW (lo16 in w1 hi)
        g1[2] = (int)((unsigned)Cin << 16); // tensor_dim0 hi=0 | tensor_dim1=Cin
        g1[3] = (int)(64u << 16);           // tensor_dim1 hi=0 | tile_dim0=64
        g1[4] = 32;                         // tile_dim1=32, tile_dim2=0
        g1[5] = HW;                         // tensor_dim0_stride lo32
        g1[6] = 0;                          // stride hi, dim1_stride lo
        g1[7] = 0;
        v4i z4; z4[0] = 0; z4[1] = 0; z4[2] = 0; z4[3] = 0;
        v8i z8;
        #pragma unroll
        for (int i = 0; i < 8; ++i) z8[i] = 0;
        __builtin_amdgcn_tensor_load_to_lds(g0, g1, z4, z4, z8, 0);
    };

    int cur = 0;
    if (wave == 0) tdm_issue(0, ldsX0);

    for (int k0 = 0; k0 < Cin; k0 += 32) {
        __syncthreads();  // prior reads of sW and of the buffer we overwrite
        const bool more = (k0 + 32) < Cin;
        if (wave == 0 && more) tdm_issue(k0 + 32, cur ? ldsX0 : ldsX1);

        {   // stage weights fragment-major: dest[m][half*16+i] for k -> (half,i)
            const int m = t >> 1, kb = (t & 1) << 4;
            const float* src = W + (long)(co0 + m) * Cin + k0 + kb;
            #pragma unroll
            for (int j = 0; j < 16; j += 2) {
                const int k  = kb + j;
                const int kk = (k < 16) ? k : (k - 16);
                const int hf = kk >> 3;
                const int ii = (kk & 7) + ((k < 16) ? 0 : 8);
                const unsigned lo = f2bf_bits(src[j]);
                const unsigned hi = f2bf_bits(src[j + 1]);
                *(unsigned*)(&sW[m * SW_LD + hf * 16 + ii]) = lo | (hi << 16);
            }
        }
        if (wave == 0) {
            if (more) __builtin_amdgcn_s_wait_tensorcnt(1);  // current tile done
            else      __builtin_amdgcn_s_wait_tensorcnt(0);  // drain
        }
        __syncthreads();

        // A fragment: two ds_load_b128 (fragment-major rows, 80B stride)
        union { uint4v q[2]; bf16x16 v; } au;
        const __bf16* arow = &sW[(wave * 16 + l15) * SW_LD + half * 16];
        au.q[0] = *(const uint4v*)(arow);
        au.q[1] = *(const uint4v*)(arow + 8);

        const float* xb = &sXf[cur][0];
        #pragma unroll
        for (int nn = 0; nn < 4; ++nn) {
            bf16x16 b;  // B: 32(K) x 16(N); lane = column, element -> K = a_k(i, half)
            #pragma unroll
            for (int i = 0; i < 16; ++i)
                b[i] = f2bf(xb[a_k(i, half) * 64 + nn * 16 + l15]);
            acc[nn] = __builtin_amdgcn_wmma_f32_16x16x32_bf16(
                false, au.v, false, b, (short)0, acc[nn], false, false);
        }
        cur ^= 1;
    }

    // epilogue: C/D layout -> VGPR r: lanes0-15 M=r, lanes16-31 M=r+8 ; N = l15
    #pragma unroll
    for (int nn = 0; nn < 4; ++nn) {
        #pragma unroll
        for (int r = 0; r < 8; ++r) {
            const int co = co0 + wave * 16 + r + half * 8;
            const int p  = p0 + nn * 16 + l15;
            float v = acc[nn][r] + bias[co];
            if (FUSE) {
                const float s = rsqrtf(bn_v[co] + 1e-5f) * bn_g[co];
                v = v * s + (bn_b[co] - bn_m[co] * s);
                v = gelu_exact(v);
            }
            Y[(long)n * yBS + (long)(yCoff + co) * HW + p] = v;
        }
    }
}

// ---------------------------------------------------------------------------
// Depthwise 3x3, pad=1, optional fused BN+GELU
// ---------------------------------------------------------------------------
template <bool FUSE>
__global__ __launch_bounds__(256)
void dw3x3(const float* __restrict__ in, long inBS,
           const float* __restrict__ w, const float* __restrict__ b,
           const float* __restrict__ bn_g, const float* __restrict__ bn_b,
           const float* __restrict__ bn_m, const float* __restrict__ bn_v,
           float* __restrict__ out, long outBS)
{
    const long idx = (long)blockIdx.x * 256 + threadIdx.x;  // NB*256*HW
    const int p = (int)(idx & (HW - 1));
    const int c = (int)((idx >> 14) & 255);
    const int n = (int)(idx >> 22);
    const int h = p >> 7, x = p & (IMW - 1);

    const float* ip = in + (long)n * inBS + (long)c * HW;
    const float* wc = w + c * 9;
    float acc = b[c];
    #pragma unroll
    for (int dy = -1; dy <= 1; ++dy) {
        const int hh = h + dy;
        if (hh < 0 || hh >= IMW) continue;
        #pragma unroll
        for (int dx = -1; dx <= 1; ++dx) {
            const int xx = x + dx;
            if (xx < 0 || xx >= IMW) continue;
            acc += wc[(dy + 1) * 3 + (dx + 1)] * ip[hh * IMW + xx];
        }
    }
    if (FUSE) {
        const float s = rsqrtf(bn_v[c] + 1e-5f) * bn_g[c];
        acc = acc * s + (bn_b[c] - bn_m[c] * s);
        acc = gelu_exact(acc);
    }
    out[(long)n * outBS + (long)c * HW + p] = acc;
}

// ---------------------------------------------------------------------------
// Windowed cross-attention: one wave per (batch, 4x4-window, head).
// head-dim 32 -> scores via one v_wmma_f32_16x16x32_bf16; attn@v via two
// (K=16 zero-padded to 32). Softmax done in C-layout registers (16-lane rows).
// ---------------------------------------------------------------------------
__global__ __launch_bounds__(256)
void win_attn(const float* __restrict__ Q, const float* __restrict__ K,
              const float* __restrict__ V, float scale, float* __restrict__ O)
{
    __shared__ float sprob[8 * 256];  // per-wave 16x16 prob tile

    const int wave = threadIdx.x >> 5, lane = threadIdx.x & 31;
    const int half = lane >> 4, l15 = lane & 15;
    const int task = blockIdx.x * 8 + wave;          // NB*1024*8 = 32768
    const int hd  = task & 7;
    const int win = (task >> 3) & 1023;
    const int n   = task >> 13;
    const int hb = win >> 5, wb = win & 31;          // 32x32 windows of 4x4

    const long base = (long)n * (256L * HW) + (long)hd * 32 * HW;

    // token m -> pixel (row-major inside window)
    #define TOKPIX(m) ((hb * 4 + ((m) >> 2)) * IMW + (wb * 4 + ((m) & 3)))

    // q fragment: A 16x32 (row = token, K = channel within head)
    bf16x16 a;
    {
        const int pq = TOKPIX(l15);
        #pragma unroll
        for (int i = 0; i < 16; ++i)
            a[i] = f2bf(Q[base + (long)a_k(i, half) * HW + pq]);
    }
    // k fragment: B 32x16, B[c][j] = k[j][c]  (lane = key token j)
    bf16x16 bk;
    {
        const int pk = TOKPIX(l15);
        #pragma unroll
        for (int i = 0; i < 16; ++i)
            bk[i] = f2bf(K[base + (long)a_k(i, half) * HW + pk]);
    }
    floatx8 s;
    #pragma unroll
    for (int i = 0; i < 8; ++i) s[i] = 0.0f;
    s = __builtin_amdgcn_wmma_f32_16x16x32_bf16(false, a, false, bk, (short)0, s, false, false);

    // softmax over keys: row M = r + half*8 lives across this 16-lane group
    #pragma unroll
    for (int r = 0; r < 8; ++r) {
        float v = s[r] * scale;
        float mx = v;
        #pragma unroll
        for (int m = 8; m >= 1; m >>= 1) mx = fmaxf(mx, __shfl_xor(mx, m, 32));
        float e = __expf(v - mx);
        float sm = e;
        #pragma unroll
        for (int m = 8; m >= 1; m >>= 1) sm += __shfl_xor(sm, m, 32);
        sprob[wave * 256 + (r + half * 8) * 16 + l15] = e / sm;
    }
    __syncthreads();

    // attn fragment: A 16x32, K-dim = key token (16 valid, upper 16 zero)
    bf16x16 pa;
    #pragma unroll
    for (int i = 0; i < 16; ++i) {
        const int k = a_k(i, half);
        pa[i] = (k < 16) ? f2bf(sprob[wave * 256 + l15 * 16 + k]) : f2bf(0.0f);
    }
    // out = attn @ v : N = 32 channels -> two 16-wide tiles
    #pragma unroll
    for (int tile = 0; tile < 2; ++tile) {
        bf16x16 bv;
        #pragma unroll
        for (int i = 0; i < 16; ++i) {
            const int kj = a_k(i, half);   // key token index
            bv[i] = (kj < 16)
                  ? f2bf(V[base + (long)(tile * 16 + l15) * HW + TOKPIX(kj)])
                  : f2bf(0.0f);
        }
        floatx8 o;
        #pragma unroll
        for (int i = 0; i < 8; ++i) o[i] = 0.0f;
        o = __builtin_amdgcn_wmma_f32_16x16x32_bf16(false, pa, false, bv, (short)0, o, false, false);
        #pragma unroll
        for (int r = 0; r < 8; ++r) {
            const int tok = r + half * 8;
            O[base + (long)(tile * 16 + l15) * HW + TOKPIX(tok)] = o[r];
        }
    }
    #undef TOKPIX
}

// ---------------------------------------------------------------------------
// Global average pool over HW: one block per (n,c)
// ---------------------------------------------------------------------------
__global__ __launch_bounds__(256)
void mean_hw(const float* __restrict__ in, float* __restrict__ out)
{
    __shared__ float red[256];
    const int nc = blockIdx.x;
    const float* p = in + (long)nc * HW;
    float s = 0.0f;
    for (int i = threadIdx.x; i < HW; i += 256) s += p[i];
    red[threadIdx.x] = s;
    __syncthreads();
    for (int off = 128; off > 0; off >>= 1) {
        if (threadIdx.x < off) red[threadIdx.x] += red[threadIdx.x + off];
        __syncthreads();
    }
    if (threadIdx.x == 0) out[nc] = red[0] * (1.0f / HW);
}

// gate[n][co] = mean[n] . Wl[co] + bl[co]   (Wl row-major (256,256))
__global__ __launch_bounds__(256)
void gate_linear(const float* __restrict__ mean, const float* __restrict__ Wl,
                 const float* __restrict__ bl, float* __restrict__ g)
{
    const int n = blockIdx.x, co = threadIdx.x;
    const float* m = mean + n * 256;
    const float* w = Wl + co * 256;
    float s = bl[co];
    for (int ci = 0; ci < 256; ++ci) s += m[ci] * w[ci];
    g[n * 256 + co] = s;
}

// write xg, yg, xg*yg into CAT channel blocks 0/1/2 (CAT is (NB,1024,HW))
__global__ __launch_bounds__(256)
void gate_apply(const float* __restrict__ XT, const float* __restrict__ Yin,
                const float* __restrict__ xw, const float* __restrict__ yw,
                float* __restrict__ CAT)
{
    const long idx = (long)blockIdx.x * 256 + threadIdx.x;  // NB*256*HW
    const int p = (int)(idx & (HW - 1));
    const int c = (int)((idx >> 14) & 255);
    const int n = (int)(idx >> 22);
    const float xv = XT[idx] * xw[n * 256 + c];
    const float yv = Yin[idx] * yw[n * 256 + c];
    const long cb = (long)n * (1024L * HW) + (long)c * HW + p;
    CAT[cb]               = xv;
    CAT[cb + 256L * HW]   = yv;
    CAT[cb + 512L * HW]   = xv * yv;
}

// ---------------------------------------------------------------------------
// Host-side orchestration
// ---------------------------------------------------------------------------
// params flat index map (setup_inputs insertion order)
enum {
    P_X = 0, P_Y, P_TRANS_W, P_TRANS_B, P_LI1_W, P_LI1_B, P_LI2_W, P_LI2_B,
    P_QX = 8, P_KX = 12, P_VX = 16, P_PROJX = 20,
    P_QY = 24, P_KY = 28, P_VY = 32, P_PROJY = 36, P_F2 = 40,
    P_CONCAT_W = 44, P_CONCAT_B, P_F1_PW_W, P_F1_PW_B, P_F1_DW_W, P_F1_DW_B,
    P_F3_W, P_F3_B,
    P_BN1 = 52, P_BN2 = 56, P_BN3 = 60
};

extern "C" void kernel_launch(void* const* d_in, const int* in_sizes, int n_in,
                              void* d_out, int out_size, void* d_ws, size_t ws_size,
                              hipStream_t stream)
{
    (void)in_sizes; (void)n_in; (void)out_size; (void)ws_size;
    const float* F[64];
    for (int i = 0; i < 64; ++i) F[i] = (const float*)d_in[i];

    float* ws = (float*)d_ws;
    const long TS = (long)NB * 256 * HW;           // 16.78M floats
    float* CAT  = ws;                              // NB x 1024 x HW
    float* ATT2 = CAT  + (long)NB * 1024 * HW;     // NB x 512  x HW
    float* XT   = ATT2 + (long)NB * 512 * HW;      // temp (also reused as E)
    float* TA   = XT  + TS;
    float* Qb   = TA  + TS;
    float* Kb   = Qb  + TS;
    float* Vb   = Kb  + TS;
    float* xmean = Vb + TS;                        // 4*256 each
    float* ymean = xmean + NB * 256;
    float* xg    = ymean + NB * 256;
    float* yg    = xg    + NB * 256;

    const dim3 gBlk(256);
    const dim3 gGemm(HW / 64, 256 / 128, NB);      // (256, 2, 4)
    const int  gElem = (int)(TS / 256);            // 65536 blocks
    const int  gAttn = (NB * 1024 * 8) / 8;        // 4096 blocks
    const long BS256  = 256L * HW;
    const long BS512  = 512L * HW;
    const long BS1024 = 1024L * HW;
    const float* Z = nullptr;

    // 1) xt = pw(x, trans)
    pw_gemm<false><<<gGemm, gBlk, 0, stream>>>(F[P_X], BS512, F[P_TRANS_W], F[P_TRANS_B],
                                               512, Z, Z, Z, Z, XT, BS256, 0);
    // 2) pooled means + gates
    mean_hw<<<NB * 256, gBlk, 0, stream>>>(XT, xmean);
    mean_hw<<<NB * 256, gBlk, 0, stream>>>(F[P_Y], ymean);
    gate_linear<<<NB, gBlk, 0, stream>>>(xmean, F[P_LI1_W], F[P_LI1_B], xg);
    gate_linear<<<NB, gBlk, 0, stream>>>(ymean, F[P_LI2_W], F[P_LI2_B], yg);
    // 3) xg/yg/out1 into CAT blocks 0..2
    gate_apply<<<gElem, gBlk, 0, stream>>>(XT, F[P_Y], xg, yg, CAT);

    const float* XG = CAT;               // CAT block 0 (stride 1024*HW per batch)
    const float* YG = CAT + 256L * HW;   // CAT block 1

    // depthwise-separable helper (dw then pw)
    auto dsc = [&](const float* src, long srcBS, int pidx, float* dst, long dstBS, int dstCoff) {
        dw3x3<false><<<gElem, gBlk, 0, stream>>>(src, srcBS, F[pidx], F[pidx + 1],
                                                 Z, Z, Z, Z, TA, BS256);
        pw_gemm<false><<<gGemm, gBlk, 0, stream>>>(TA, BS256, F[pidx + 2], F[pidx + 3],
                                                   256, Z, Z, Z, Z, dst, dstBS, dstCoff);
    };

    const float scale = 0.0625f;  // 256^-0.5

    // 4) attnx = cross_attn(q=yg, k=xg, v=xg ; qy,kx,vx,projx) -> ATT2[:,0:256]
    dsc(YG, BS1024, P_QY, Qb, BS256, 0);
    dsc(XG, BS1024, P_KX, Kb, BS256, 0);
    dsc(XG, BS1024, P_VX, Vb, BS256, 0);
    win_attn<<<gAttn, gBlk, 0, stream>>>(Qb, Kb, Vb, scale, XT);  // XT reused
    dw3x3<false><<<gElem, gBlk, 0, stream>>>(XT, BS256, F[P_PROJX], F[P_PROJX + 1],
                                             Z, Z, Z, Z, TA, BS256);
    pw_gemm<false><<<gGemm, gBlk, 0, stream>>>(TA, BS256, F[P_PROJX + 2], F[P_PROJX + 3],
                                               256, Z, Z, Z, Z, ATT2, BS512, 0);

    // 5) attny = cross_attn(q=xg, k=yg, v=yg ; qx,ky,vy,projy) -> ATT2[:,256:512]
    dsc(XG, BS1024, P_QX, Qb, BS256, 0);
    dsc(YG, BS1024, P_KY, Kb, BS256, 0);
    dsc(YG, BS1024, P_VY, Vb, BS256, 0);
    win_attn<<<gAttn, gBlk, 0, stream>>>(Qb, Kb, Vb, scale, XT);
    dw3x3<false><<<gElem, gBlk, 0, stream>>>(XT, BS256, F[P_PROJY], F[P_PROJY + 1],
                                             Z, Z, Z, Z, TA, BS256);
    pw_gemm<false><<<gGemm, gBlk, 0, stream>>>(TA, BS256, F[P_PROJY + 2], F[P_PROJY + 3],
                                               256, Z, Z, Z, Z, ATT2, BS512, 256);

    // 6) out2 = pw(concat(attnx,attny), concat_w) -> CAT block 3
    pw_gemm<false><<<gGemm, gBlk, 0, stream>>>(ATT2, BS512, F[P_CONCAT_W], F[P_CONCAT_B],
                                               512, Z, Z, Z, Z, CAT, BS1024, 768);

    // 7) f1 (inverted dsc: pw 1024->256 then dw) + BN1 + GELU
    pw_gemm<false><<<gGemm, gBlk, 0, stream>>>(CAT, BS1024, F[P_F1_PW_W], F[P_F1_PW_B],
                                               1024, Z, Z, Z, Z, TA, BS256, 0);
    dw3x3<true><<<gElem, gBlk, 0, stream>>>(TA, BS256, F[P_F1_DW_W], F[P_F1_DW_B],
                                            F[P_BN1], F[P_BN1 + 1], F[P_BN1 + 2], F[P_BN1 + 3],
                                            XT, BS256);

    // 8) f2 (dsc: dw then pw) + BN2 + GELU
    dw3x3<false><<<gElem, gBlk, 0, stream>>>(XT, BS256, F[P_F2], F[P_F2 + 1],
                                             Z, Z, Z, Z, TA, BS256);
    pw_gemm<true><<<gGemm, gBlk, 0, stream>>>(TA, BS256, F[P_F2 + 2], F[P_F2 + 3], 256,
                                              F[P_BN2], F[P_BN2 + 1], F[P_BN2 + 2], F[P_BN2 + 3],
                                              XT, BS256, 0);

    // 9) f3 pw + BN3 + GELU -> output
    pw_gemm<true><<<gGemm, gBlk, 0, stream>>>(XT, BS256, F[P_F3_W], F[P_F3_B], 256,
                                              F[P_BN3], F[P_BN3 + 1], F[P_BN3 + 2], F[P_BN3 + 3],
                                              (float*)d_out, BS256, 0);
}
